// LSTM_49203145343583
// MI455X (gfx1250) — compile-verified, matrix operand order checked by reading, not run on hardware
//
#include <hip/hip_runtime.h>
#include <hip/hip_bf16.h>
#include <stdint.h>

// ---------------------------------------------------------------------------
// LSTM LM on MI455X (gfx1250): bf16 WMMA GEMMs + fused per-timestep recurrence
//   B=64, T=128, V=10000, E=H=1024, L=2
// Software-pipelined (register double-buffered) WMMA loops so global loads
// overlap WMMA issue instead of s_wait_loadcnt 0x0 before every WMMA.
// ---------------------------------------------------------------------------

typedef __attribute__((ext_vector_type(16))) __bf16 v16bf;
typedef __attribute__((ext_vector_type(8)))  float  v8f;

static constexpr int kB = 64, kT = 128, kV = 10000, kE = 1024, kH = 1024, kL = 2;
static constexpr int kBT = kB * kT;                   // 8192
static constexpr size_t kLogits = (size_t)kBT * kV;   // 81,920,000 floats

union ABTile { v16bf v; uint32_t u[8]; uint4 q[2]; };

// Load one 16x32 bf16 tile (A, or B given as W^T rows) from row-major [rows,ld].
// ISA 16-bit A 16x32 layout: lanes 0-15 = rows 0-15 with K=0..7,16..23;
// lanes 16-31 same rows with K=8..15,24..31.  Two contiguous 16B loads/lane.
__device__ __forceinline__ v16bf load_tile16x32(const __bf16* __restrict__ base,
                                                int ld, int r0, int k0) {
  int lane = threadIdx.x & 31;
  int r  = r0 + (lane & 15);
  int kb = k0 + ((lane >> 4) << 3);
  const __bf16* p = base + (size_t)r * ld + kb;
  ABTile t;
  t.q[0] = *(const uint4*)(p);
  t.q[1] = *(const uint4*)(p + 16);
  return t.v;
}

__device__ __forceinline__ float sigmoid_fast(float x) {
  return 1.0f / (1.0f + __expf(-x));
}
__device__ __forceinline__ float tanh_fast(float x) {
  x = fminf(fmaxf(x, -15.0f), 15.0f);
  float e = __expf(2.0f * x);
  return (e - 1.0f) / (e + 1.0f);
}

#define WMMA_BF16(A_, B_, C_) \
  __builtin_amdgcn_wmma_f32_16x16x32_bf16(false, (A_), false, (B_), (short)0, (C_), false, false)

// ---------------------------------------------------------------------------
// C[M,N] = A[M,K] * W[N,K]^T + bias[N].  A,W row-major bf16, C f32.
// One wave per block, 32(M) x 64(N): 8 accumulators, A x4 / B x2 reuse.
// Register double-buffered K loop.  NO bounds checks (exact tiling).
// NT: nontemporal stores for write-once outputs (logits).
// ---------------------------------------------------------------------------
template<bool NT>
__global__ void __launch_bounds__(32)
gemm_bf16_nt32x64(const __bf16* __restrict__ A, const __bf16* __restrict__ W,
                  const float* __restrict__ bias, float* __restrict__ C,
                  int N, int K, int nofs) {
  const int m0    = blockIdx.x * 32;
  const int nbase = nofs + blockIdx.y * 64;
  v8f c00 = {}, c01 = {}, c02 = {}, c03 = {};
  v8f c10 = {}, c11 = {}, c12 = {}, c13 = {};

  // Prologue: tiles for k0 = 0
  v16bf a0 = load_tile16x32(A, K, m0,      0);
  v16bf a1 = load_tile16x32(A, K, m0 + 16, 0);
  v16bf b0 = load_tile16x32(W, K, nbase +  0, 0);
  v16bf b1 = load_tile16x32(W, K, nbase + 16, 0);
  v16bf b2 = load_tile16x32(W, K, nbase + 32, 0);
  v16bf b3 = load_tile16x32(W, K, nbase + 48, 0);

  for (int k0 = 32; k0 < K; k0 += 32) {
    // Prefetch next K-slice into fresh registers (overlaps WMMAs below).
    v16bf na0 = load_tile16x32(A, K, m0,      k0);
    v16bf na1 = load_tile16x32(A, K, m0 + 16, k0);
    v16bf nb0 = load_tile16x32(W, K, nbase +  0, k0);
    v16bf nb1 = load_tile16x32(W, K, nbase + 16, k0);
    v16bf nb2 = load_tile16x32(W, K, nbase + 32, k0);
    v16bf nb3 = load_tile16x32(W, K, nbase + 48, k0);

    c00 = WMMA_BF16(a0, b0, c00);  c10 = WMMA_BF16(a1, b0, c10);
    c01 = WMMA_BF16(a0, b1, c01);  c11 = WMMA_BF16(a1, b1, c11);
    c02 = WMMA_BF16(a0, b2, c02);  c12 = WMMA_BF16(a1, b2, c12);
    c03 = WMMA_BF16(a0, b3, c03);  c13 = WMMA_BF16(a1, b3, c13);

    a0 = na0; a1 = na1; b0 = nb0; b1 = nb1; b2 = nb2; b3 = nb3;
  }
  // Tail
  c00 = WMMA_BF16(a0, b0, c00);  c10 = WMMA_BF16(a1, b0, c10);
  c01 = WMMA_BF16(a0, b1, c01);  c11 = WMMA_BF16(a1, b1, c11);
  c02 = WMMA_BF16(a0, b2, c02);  c12 = WMMA_BF16(a1, b2, c12);
  c03 = WMMA_BF16(a0, b3, c03);  c13 = WMMA_BF16(a1, b3, c13);

  const int lane = threadIdx.x & 31;
  const int nl   = lane & 15;
  const int mhi  = (lane >> 4) << 3;   // C/D layout: VGPR e -> M = e + 8*(lane/16)
  v8f accs[2][4] = {{c00, c01, c02, c03}, {c10, c11, c12, c13}};
#pragma unroll
  for (int r = 0; r < 2; ++r) {
#pragma unroll
    for (int j = 0; j < 4; ++j) {
      int n = nbase + 16 * j + nl;
      float bb = bias[n];
#pragma unroll
      for (int e = 0; e < 8; ++e) {
        int m = m0 + 16 * r + e + mhi;
        float v = accs[r][j][e] + bb;
        if (NT) __builtin_nontemporal_store(v, &C[(size_t)m * N + n]);
        else    C[(size_t)m * N + n] = v;
      }
    }
  }
}

// Edge strip: 32(M) x 16(N) per wave at fixed column offset (still unguarded:
// V = 10000 is an exact multiple of 16).
template<bool NT>
__global__ void __launch_bounds__(32)
gemm_bf16_nt32x16(const __bf16* __restrict__ A, const __bf16* __restrict__ W,
                  const float* __restrict__ bias, float* __restrict__ C,
                  int N, int K, int nofs) {
  const int m0 = blockIdx.x * 32;
  const int n0 = nofs;
  v8f acc0 = {}, acc1 = {};
  v16bf a0 = load_tile16x32(A, K, m0,      0);
  v16bf a1 = load_tile16x32(A, K, m0 + 16, 0);
  v16bf b  = load_tile16x32(W, K, n0,      0);
  for (int k0 = 32; k0 < K; k0 += 32) {
    v16bf na0 = load_tile16x32(A, K, m0,      k0);
    v16bf na1 = load_tile16x32(A, K, m0 + 16, k0);
    v16bf nb  = load_tile16x32(W, K, n0,      k0);
    acc0 = WMMA_BF16(a0, b, acc0);
    acc1 = WMMA_BF16(a1, b, acc1);
    a0 = na0; a1 = na1; b = nb;
  }
  acc0 = WMMA_BF16(a0, b, acc0);
  acc1 = WMMA_BF16(a1, b, acc1);

  const int lane = threadIdx.x & 31;
  const int n    = n0 + (lane & 15);
  const int mhi  = (lane >> 4) << 3;
  float bb = bias[n];
#pragma unroll
  for (int e = 0; e < 8; ++e) {
    float v0 = acc0[e] + bb, v1 = acc1[e] + bb;
    size_t i0 = (size_t)(m0 + e + mhi) * N + n;
    size_t i1 = (size_t)(m0 + 16 + e + mhi) * N + n;
    if (NT) { __builtin_nontemporal_store(v0, &C[i0]); __builtin_nontemporal_store(v1, &C[i1]); }
    else    { C[i0] = v0; C[i1] = v1; }
  }
}

// ---------------------------------------------------------------------------
// Fused LSTM timestep: gates = gx[:,t,:] + h @ Wh^T, then pointwise update.
// Grid: (B/16, H/16) waves; each wave computes the SAME 16x16 (batch x hcol)
// tile of all four gates -> epilogue is purely register-elementwise.
// Register double-buffered K loop to overlap Wh/h loads with WMMAs (this is
// the serialized critical path of the whole model).
// ---------------------------------------------------------------------------
__global__ void __launch_bounds__(32)
lstm_step(const float* __restrict__ gx,       // [B*T, 4H] (bias folded in)
          const __bf16* __restrict__ wh,      // [4H, H] bf16
          const int* __restrict__ lengths,    // [B]
          float* __restrict__ h, float* __restrict__ c,
          __bf16* __restrict__ hbf,           // [B, H] bf16 copy of h
          __bf16* __restrict__ xout,          // [B*T, H] layer output (bf16)
          int t) {
  const int m0 = blockIdx.x * 16;   // batch tile
  const int n0 = blockIdx.y * 16;   // hidden-column tile
  v8f ai = {}, af = {}, ag = {}, ao = {};

  const __bf16* whI = wh;
  const __bf16* whF = wh + (size_t)1 * kH * kH;
  const __bf16* whG = wh + (size_t)2 * kH * kH;
  const __bf16* whO = wh + (size_t)3 * kH * kH;

  v16bf a  = load_tile16x32(hbf, kH, m0, 0);
  v16bf bi = load_tile16x32(whI, kH, n0, 0);
  v16bf bf_ = load_tile16x32(whF, kH, n0, 0);
  v16bf bg = load_tile16x32(whG, kH, n0, 0);
  v16bf bo = load_tile16x32(whO, kH, n0, 0);

  for (int k0 = 32; k0 < kH; k0 += 32) {
    v16bf na  = load_tile16x32(hbf, kH, m0, k0);
    v16bf nbi = load_tile16x32(whI, kH, n0, k0);
    v16bf nbf = load_tile16x32(whF, kH, n0, k0);
    v16bf nbg = load_tile16x32(whG, kH, n0, k0);
    v16bf nbo = load_tile16x32(whO, kH, n0, k0);

    ai = WMMA_BF16(a, bi,  ai);
    af = WMMA_BF16(a, bf_, af);
    ag = WMMA_BF16(a, bg,  ag);
    ao = WMMA_BF16(a, bo,  ao);

    a = na; bi = nbi; bf_ = nbf; bg = nbg; bo = nbo;
  }
  ai = WMMA_BF16(a, bi,  ai);
  af = WMMA_BF16(a, bf_, af);
  ag = WMMA_BF16(a, bg,  ag);
  ao = WMMA_BF16(a, bo,  ao);

  const int lane = threadIdx.x & 31;
  const int n    = n0 + (lane & 15);
  const int mhi  = (lane >> 4) << 3;
#pragma unroll
  for (int e = 0; e < 8; ++e) {
    int b = m0 + e + mhi;                          // batch row
    size_t grow = ((size_t)(b * kT + t)) * (4 * kH);
    float iv = ai[e] + gx[grow + 0 * kH + n];
    float fv = af[e] + gx[grow + 1 * kH + n];
    float gv = ag[e] + gx[grow + 2 * kH + n];
    float ov = ao[e] + gx[grow + 3 * kH + n];
    size_t hc = (size_t)b * kH + n;
    float c_old = c[hc];
    float cn = sigmoid_fast(fv) * c_old + sigmoid_fast(iv) * tanh_fast(gv);
    float hn = sigmoid_fast(ov) * tanh_fast(cn);
    bool live = t < lengths[b];                    // packed-seq freeze
    if (live) {
      c[hc]   = cn;
      h[hc]   = hn;
      hbf[hc] = (__bf16)hn;
    }
    xout[((size_t)(b * kT + t)) * kH + n] = (__bf16)(live ? hn : 0.0f);
  }
}

// ---------------------------------------------------------------------------
// Small helpers
// ---------------------------------------------------------------------------
__global__ void compute_lengths(const int* __restrict__ mask, int* __restrict__ len) {
  int b = threadIdx.x;
  if (b < kB) {
    int s = 0;
    for (int t = 0; t < kT; ++t) s += mask[b * kT + t];
    len[b] = s;
  }
}

__global__ void bias_sum_k(const float* __restrict__ bi, const float* __restrict__ bh,
                           float* __restrict__ o, int n) {
  int i = blockIdx.x * blockDim.x + threadIdx.x;
  if (i < n) o[i] = bi[i] + bh[i];
}

__global__ void f32_to_bf16_k(const float* __restrict__ s, __bf16* __restrict__ d, size_t n) {
  size_t i = (size_t)blockIdx.x * blockDim.x + threadIdx.x;
  if (i < n) d[i] = (__bf16)s[i];
}

__global__ void gather_embed(const int* __restrict__ ids, const float* __restrict__ emb,
                             __bf16* __restrict__ x) {
  int row = blockIdx.x;                  // 0..B*T-1
  int id  = ids[row];
  const float* src = emb + (size_t)id * kE;
  __bf16* dst = x + (size_t)row * kE;
  for (int j = threadIdx.x; j < kE; j += blockDim.x) dst[j] = (__bf16)src[j];
}

__global__ void fill_zero_u32(uint32_t* __restrict__ p, size_t n) {
  size_t i = (size_t)blockIdx.x * blockDim.x + threadIdx.x;
  if (i < n) p[i] = 0u;
}

__global__ void copy_f32(const float* __restrict__ s, float* __restrict__ d, size_t n) {
  size_t i = (size_t)blockIdx.x * blockDim.x + threadIdx.x;
  if (i < n) d[i] = s[i];
}

// ---------------------------------------------------------------------------
// kernel_launch
// ---------------------------------------------------------------------------
extern "C" void kernel_launch(void* const* d_in, const int* in_sizes, int n_in,
                              void* d_out, int out_size, void* d_ws, size_t ws_size,
                              hipStream_t stream) {
  const int*   ids   = (const int*)d_in[0];
  const int*   amask = (const int*)d_in[1];
  const float* emb   = (const float*)d_in[2];
  const float* W_ih  = (const float*)d_in[3];   // [L, 4H, E]
  const float* W_hh  = (const float*)d_in[4];   // [L, 4H, H]
  const float* b_ih  = (const float*)d_in[5];   // [L, 4H]
  const float* b_hh  = (const float*)d_in[6];
  const float* fc_w  = (const float*)d_in[7];   // [V, H]
  const float* fc_b  = (const float*)d_in[8];   // [V]
  float* out = (float*)d_out;

  char* ws = (char*)d_ws;
  size_t off = 0;
  auto alloc = [&](size_t bytes) -> void* {
    void* p = ws + off;
    off += (bytes + 255) & ~(size_t)255;
    return p;
  };
  int*    len  = (int*)    alloc((size_t)kB * 4);
  float*  bsum = (float*)  alloc((size_t)kL * 4 * kH * 4);
  __bf16* xbf  = (__bf16*) alloc((size_t)kBT * kE * 2);        // activations (bf16)
  float*  gx   = (float*)  alloc((size_t)kBT * 4 * kH * 4);    // input projection
  __bf16* wi   = (__bf16*) alloc((size_t)4 * kH * kE * 2);     // per-layer Wi bf16
  __bf16* wh   = (__bf16*) alloc((size_t)4 * kH * kH * 2);     // per-layer Wh bf16
  __bf16* fcw  = (__bf16*) alloc((size_t)kV * kH * 2);         // FC weight bf16
  float*  h    = (float*)  alloc((size_t)kB * kH * 4);
  float*  c    = (float*)  alloc((size_t)kB * kH * 4);
  __bf16* hbf  = (__bf16*) alloc((size_t)kB * kH * 2);

  // Preliminaries
  compute_lengths<<<1, 64, 0, stream>>>(amask, len);
  bias_sum_k<<<(kL * 4 * kH + 255) / 256, 256, 0, stream>>>(b_ih, b_hh, bsum, kL * 4 * kH);
  gather_embed<<<kBT, 256, 0, stream>>>(ids, emb, xbf);
  {
    size_t n = (size_t)kV * kH;
    f32_to_bf16_k<<<(unsigned)((n + 255) / 256), 256, 0, stream>>>(fc_w, fcw, n);
  }

  for (int l = 0; l < kL; ++l) {
    // Weights for this layer -> bf16
    size_t nw = (size_t)4 * kH * kE;
    f32_to_bf16_k<<<(unsigned)((nw + 255) / 256), 256, 0, stream>>>(W_ih + (size_t)l * nw, wi, nw);
    f32_to_bf16_k<<<(unsigned)((nw + 255) / 256), 256, 0, stream>>>(W_hh + (size_t)l * nw, wh, nw);

    // gx = x @ Wi^T + (b_ih + b_hh)   [8192 x 4096], exact 32x64 tiling
    dim3 gg(kBT / 32, (4 * kH) / 64);
    gemm_bf16_nt32x64<false><<<gg, 32, 0, stream>>>(xbf, wi, bsum + (size_t)l * 4 * kH, gx,
                                                    4 * kH, kE, 0);

    // h = c = 0
    fill_zero_u32<<<(kB * kH + 255) / 256, 256, 0, stream>>>((uint32_t*)h, (size_t)kB * kH);
    fill_zero_u32<<<(kB * kH + 255) / 256, 256, 0, stream>>>((uint32_t*)c, (size_t)kB * kH);
    fill_zero_u32<<<(kB * kH / 2 + 255) / 256, 256, 0, stream>>>((uint32_t*)hbf, (size_t)kB * kH / 2);

    // Sequential recurrence; stream order carries the h -> h dependency.
    dim3 gs(kB / 16, kH / 16);
    for (int t = 0; t < kT; ++t)
      lstm_step<<<gs, 32, 0, stream>>>(gx, wh, len, h, c, hbf, xbf, t);

    // Final states into output tuple slots
    copy_f32<<<(kB * kH + 255) / 256, 256, 0, stream>>>(h, out + kLogits + (size_t)l * kB * kH,
                                                        (size_t)kB * kH);
    copy_f32<<<(kB * kH + 255) / 256, 256, 0, stream>>>(c, out + kLogits + (size_t)kL * kB * kH + (size_t)l * kB * kH,
                                                        (size_t)kB * kH);
  }

  // logits = y @ fc_w^T + fc_b   [8192 x 10000]
  // Main body: 156 groups of 64 columns (9984); edge: one 16-column strip.
  dim3 gf(kBT / 32, (kV / 64));                 // 156 full 64-col groups
  gemm_bf16_nt32x64<true><<<gf, 32, 0, stream>>>(xbf, fcw, fc_b, out, kV, kE, 0);
  gemm_bf16_nt32x16<true><<<dim3(kBT / 32, 1), 32, 0, stream>>>(xbf, fcw, fc_b, out,
                                                                kV, kE, (kV / 64) * 64);
}
